// DynamicMoEModel_33689723470017
// MI455X (gfx1250) — matrix operand order repaired and checked by reading, not compiled, not run on hardware
//
#include <hip/hip_runtime.h>
#include <hip/hip_bf16.h>

// ---------------- types ----------------
typedef __bf16 bf16_t;
typedef __attribute__((ext_vector_type(16))) __bf16 v16bf;
typedef __attribute__((ext_vector_type(8)))  __bf16 v8bf;
typedef __attribute__((ext_vector_type(8)))  float  v8f;

#define B_  4
#define S_  1024
#define V_  50257
#define H_  256
#define E_  8
#define L_  4

#define BM 128
#define BN 128
#define BK 32

// ---- CDNA5 async global->LDS copy (ASYNCcnt path, no VGPR round-trip) ----
__device__ __forceinline__ void async_copy_b128(void* lds_dst, const void* gsrc)
{
    const unsigned l = (unsigned)(uintptr_t)lds_dst;   // low 32 bits = LDS offset
    asm volatile("global_load_async_to_lds_b128 %0, %1, off"
                 :: "v"(l), "v"(gsrc) : "memory");
}
__device__ __forceinline__ void wait_async_le4()
{ asm volatile("s_wait_asynccnt 0x4" ::: "memory"); }
__device__ __forceinline__ void wait_async_0()
{ asm volatile("s_wait_asynccnt 0x0" ::: "memory"); }

// ---------------- core GEMM (bf16 WMMA, f32 acc) ----------------
// C[M,N] = A[M,K] * Bt (Bt[n][k]) + bias0(+bias1), optional ReLU.
// K may be split across two Bt pointers at k==ksplit (each row length ksplit).
// Block tile 128x128, BK=32, double-buffered LDS, async staging.
template<bool RELU, bool OUT_BF16>
__device__ __forceinline__ void gemm_core(
    const bf16_t* __restrict__ A, int lda,
    const bf16_t* __restrict__ B0t, const bf16_t* __restrict__ B1t, int ksplit,
    const float* __restrict__ bias0, const float* __restrict__ bias1,
    void* __restrict__ Cout, int ldc,
    int M, int N, int K)
{
    __shared__ bf16_t sA[2][BM][BK];   // 2 x 8 KB
    __shared__ bf16_t sB[2][BN][BK];   // 2 x 8 KB

    const int t    = threadIdx.x;
    const int lane = t & 31;
    const int wave = t >> 5;
    const int wm   = wave & 3;    // 4 wave rows (32 M each)
    const int wn   = wave >> 2;   // 2 wave cols (64 N each)
    const int m0   = blockIdx.y * BM;
    const int n0   = blockIdx.x * BN;
    const int half = lane >> 4;
    const int l16  = lane & 15;

    // per-tile staging: 4 async b128 per thread (2 for A, 2 for B)
    auto stage = [&](int kb, int buf) {
#pragma unroll
        for (int i = 0; i < 2; ++i) {
            const int c   = t + i * 256;       // 0..511
            const int row = c >> 2;            // 0..127
            const int kc  = (c & 3) << 3;      // 0,8,16,24
            async_copy_b128(&sA[buf][row][kc],
                            A + (size_t)(m0 + row) * lda + kb + kc);
        }
#pragma unroll
        for (int i = 0; i < 2; ++i) {
            const int c   = t + i * 256;
            const int row = c >> 2;
            const int kc  = (c & 3) << 3;
            int n = n0 + row;
            if (n >= N) n = N - 1;             // clamp: col n>=N never stored
            const int kg = kb + kc;
            const bf16_t* bp = (kg < ksplit)
                ? (B0t + (size_t)n * ksplit + kg)
                : (B1t + (size_t)n * ksplit + (kg - ksplit));
            async_copy_b128(&sB[buf][row][kc], bp);
        }
    };

    v8f acc[2][4] = {};

    const int nk = K / BK;
    stage(0, 0);
    for (int it = 0; it < nk; ++it) {
        const int cur = it & 1;
        if (it + 1 < nk) { stage((it + 1) * BK, cur ^ 1); wait_async_le4(); }
        else             { wait_async_0(); }
        __syncthreads();

        // fragments (ISA 16-bit A 16x32 / B 32x16 layouts)
        v16bf afr[2], bfr[4];
#pragma unroll
        for (int mt = 0; mt < 2; ++mt) {
            const int m = wm * 32 + mt * 16 + l16;
            const v8bf c0 = *(const v8bf*)&sA[cur][m][half * 8];
            const v8bf c1 = *(const v8bf*)&sA[cur][m][16 + half * 8];
#pragma unroll
            for (int i = 0; i < 8; ++i) { afr[mt][i] = c0[i]; afr[mt][8 + i] = c1[i]; }
        }
#pragma unroll
        for (int nt = 0; nt < 4; ++nt) {
            const int n = wn * 64 + nt * 16 + l16;
            const v8bf d0 = *(const v8bf*)&sB[cur][n][half * 8];
            const v8bf d1 = *(const v8bf*)&sB[cur][n][16 + half * 8];
#pragma unroll
            for (int i = 0; i < 8; ++i) { bfr[nt][i] = d0[i]; bfr[nt][8 + i] = d1[i]; }
        }

#pragma unroll
        for (int mt = 0; mt < 2; ++mt)
#pragma unroll
            for (int nt = 0; nt < 4; ++nt)
                acc[mt][nt] = __builtin_amdgcn_wmma_f32_16x16x32_bf16(
                    false, afr[mt], false, bfr[nt],
                    (short)0, acc[mt][nt], false, false);
        __syncthreads();   // protect LDS buffer reuse vs next stage()
    }

    // ---- epilogue: bias (+bias1), ReLU, store
#pragma unroll
    for (int mt = 0; mt < 2; ++mt) {
#pragma unroll
        for (int nt = 0; nt < 4; ++nt) {
            const int n = n0 + wn * 64 + nt * 16 + l16;
            if (n >= N) continue;
            float bv = 0.f;
            if (bias0) bv += bias0[n];
            if (bias1) bv += bias1[n];
            const int mbase = m0 + wm * 32 + mt * 16 + half * 8;
#pragma unroll
            for (int r = 0; r < 8; ++r) {
                float v = acc[mt][nt][r] + bv;
                if (RELU) v = v > 0.f ? v : 0.f;
                const size_t off = (size_t)(mbase + r) * ldc + n;
                if (OUT_BF16) ((bf16_t*)Cout)[off] = (bf16_t)v;
                else __builtin_nontemporal_store(v, (float*)Cout + off);
            }
        }
    }
}

// ---------------- kernels ----------------
__global__ void k_embed(const int* __restrict__ x, const float* __restrict__ emb,
                        bf16_t* __restrict__ act)
{
    const int s = blockIdx.x;        // 0..4095
    const int h = threadIdx.x;       // 0..255
    const int tok = x[s];
    act[(size_t)s * H_ + h] = (bf16_t)emb[(size_t)tok * H_ + h];
}

// Wt[e][n][k] = W[e][k][n]  (fp32 -> bf16, transposed)
__global__ void k_transpose_w(const float* __restrict__ W, bf16_t* __restrict__ Wt,
                              int Kdim, int Ndim)
{
    const size_t base = (size_t)blockIdx.z * Kdim * Ndim;
    const int k = blockIdx.y;
    const int n = blockIdx.x * 256 + threadIdx.x;
    if (n < Ndim)
        Wt[base + (size_t)n * Kdim + k] = (bf16_t)W[base + (size_t)k * Ndim + n];
}

// mean-pool over S, gate GEMV, top-2 expert selection per batch
__global__ void k_gate(const bf16_t* __restrict__ act, const float* __restrict__ gW,
                       const float* __restrict__ gb, int* __restrict__ idxOut)
{
    __shared__ float pooled[H_];
    __shared__ float logits[E_];
    const int b = blockIdx.x;
    const int h = threadIdx.x;
    float s = 0.f;
    const bf16_t* p = act + (size_t)b * S_ * H_ + h;
    for (int i = 0; i < S_; ++i) s += (float)p[(size_t)i * H_];
    pooled[h] = s * (1.0f / (float)S_);
    __syncthreads();
    if (h < E_) {
        float lg = gb[h];
        for (int i = 0; i < H_; ++i) lg += pooled[i] * gW[i * E_ + h];
        logits[h] = lg;
    }
    __syncthreads();
    if (h == 0) {
        int e0 = 0; float v0 = logits[0];
        for (int e = 1; e < E_; ++e) if (logits[e] > v0) { v0 = logits[e]; e0 = e; }
        int e1 = -1; float v1 = -3.0e38f;
        for (int e = 0; e < E_; ++e)
            if (e != e0 && logits[e] > v1) { v1 = logits[e]; e1 = e; }
        idxOut[b * 2]     = e0;
        idxOut[b * 2 + 1] = e1;
    }
}

// h[b][s][kk*256 + f] = relu(act[b] @ W1[e_{b,kk}] + b1[e]);  grid.z = B*K = 8
__global__ void k_gemm_ffn1(const bf16_t* __restrict__ act,
                            const bf16_t* __restrict__ W1t,
                            const float* __restrict__ b1,
                            const int* __restrict__ idx,
                            bf16_t* __restrict__ hbuf)
{
    const int z = blockIdx.z;
    const int b = z >> 1, kk = z & 1;
    const int e = idx[b * 2 + kk];
    gemm_core<true, true>(act + (size_t)b * S_ * H_, H_,
                          W1t + (size_t)e * H_ * H_, nullptr, H_,
                          b1 + e * H_, nullptr,
                          hbuf + (size_t)b * S_ * (2 * H_) + kk * H_, 2 * H_,
                          S_, H_, H_);
}

// actNext[b] = h[b][:, :256] @ W2[e0] + h[b][:, 256:] @ W2[e1] + b2[e0] + b2[e1]
__global__ void k_gemm_ffn2(const bf16_t* __restrict__ hbuf,
                            const bf16_t* __restrict__ W2t,
                            const float* __restrict__ b2,
                            const int* __restrict__ idx,
                            bf16_t* __restrict__ actNext)
{
    const int b  = blockIdx.z;
    const int e0 = idx[b * 2];
    const int e1 = idx[b * 2 + 1];
    gemm_core<false, true>(hbuf + (size_t)b * S_ * (2 * H_), 2 * H_,
                           W2t + (size_t)e0 * H_ * H_,
                           W2t + (size_t)e1 * H_ * H_, H_,
                           b2 + e0 * H_, b2 + e1 * H_,
                           actNext + (size_t)b * S_ * H_, H_,
                           S_, H_, 2 * H_);
}

// d_out[4096, 50257] = act @ fcW + fcb  (fp32 out, NT stores)
__global__ void k_gemm_final(const bf16_t* __restrict__ act,
                             const bf16_t* __restrict__ fcWt,
                             const float* __restrict__ fcb,
                             float* __restrict__ out)
{
    gemm_core<false, false>(act, H_,
                            fcWt, nullptr, H_,
                            fcb, nullptr,
                            out, V_,
                            B_ * S_, V_, H_);
}

// ---------------- host launch ----------------
extern "C" void kernel_launch(void* const* d_in, const int* in_sizes, int n_in,
                              void* d_out, int out_size, void* d_ws, size_t ws_size,
                              hipStream_t stream)
{
    const int*   x    = (const int*)  d_in[0];
    const float* emb  = (const float*)d_in[1];
    const float* W1   = (const float*)d_in[2];
    const float* b1   = (const float*)d_in[3];
    const float* W2   = (const float*)d_in[4];
    const float* b2   = (const float*)d_in[5];
    const float* gW   = (const float*)d_in[6];
    const float* gb   = (const float*)d_in[7];
    const float* fcW  = (const float*)d_in[8];
    const float* fcb  = (const float*)d_in[9];
    float* out = (float*)d_out;

    // workspace carve-up (all 256B aligned)
    char* ws = (char*)d_ws;
    size_t off = 0;
    auto take = [&](size_t bytes) {
        void* p = ws + off;
        off += (bytes + 255) & ~(size_t)255;
        return p;
    };
    bf16_t* actA  = (bf16_t*)take((size_t)B_ * S_ * H_ * 2);       // 2 MB
    bf16_t* actB  = (bf16_t*)take((size_t)B_ * S_ * H_ * 2);       // 2 MB
    bf16_t* hbuf  = (bf16_t*)take((size_t)B_ * S_ * 2 * H_ * 2);   // 4 MB
    bf16_t* W1t   = (bf16_t*)take((size_t)E_ * H_ * H_ * 2);       // 1 MB
    bf16_t* W2t   = (bf16_t*)take((size_t)E_ * H_ * H_ * 2);       // 1 MB
    bf16_t* fcWt  = (bf16_t*)take((size_t)V_ * H_ * 2);            // ~24.5 MB
    int*    idx   = (int*)   take((size_t)B_ * 2 * sizeof(int));
    (void)ws_size; (void)in_sizes; (void)n_in; (void)out_size;

    // weight conversion / transposition (bf16, Wt[n][k])
    k_transpose_w<<<dim3(1, H_, E_), 256, 0, stream>>>(W1, W1t, H_, H_);
    k_transpose_w<<<dim3(1, H_, E_), 256, 0, stream>>>(W2, W2t, H_, H_);
    k_transpose_w<<<dim3((V_ + 255) / 256, H_, 1), 256, 0, stream>>>(fcW, fcWt, H_, V_);

    // embedding gather -> bf16 activations
    k_embed<<<dim3(B_ * S_), 256, 0, stream>>>(x, emb, actA);

    bf16_t* cur = actA;
    bf16_t* nxt = actB;
    for (int l = 0; l < L_; ++l) {
        k_gate<<<dim3(B_), 256, 0, stream>>>(cur, gW, gb, idx);
        k_gemm_ffn1<<<dim3(H_ / BN, S_ / BM, B_ * 2), 256, 0, stream>>>(cur, W1t, b1, idx, hbuf);
        k_gemm_ffn2<<<dim3(H_ / BN, S_ / BM, B_), 256, 0, stream>>>(hbuf, W2t, b2, idx, nxt);
        bf16_t* t = cur; cur = nxt; nxt = t;
    }

    k_gemm_final<<<dim3((V_ + BN - 1) / BN, (B_ * S_) / BM, 1), 256, 0, stream>>>(cur, fcWt, fcb, out);
}
